// MoNet_62088047231392
// MI455X (gfx1250) — matrix operand order repaired and checked by reading, compiled, not run on hardware
//
#include <hip/hip_runtime.h>
#include <hip/hip_bf16.h>
#include <math.h>

typedef __attribute__((ext_vector_type(16))) _Float16 v16h;
typedef __attribute__((ext_vector_type(8)))  float    v8f;

#define HID 64
#define KK  3
#define NL  4
#define NC  7
#define BN_EPS 1e-5f

#if defined(__HIP_DEVICE_COMPILE__) && \
    __has_builtin(__builtin_amdgcn_global_load_async_to_lds_b128) && \
    __has_builtin(__builtin_amdgcn_s_wait_asynccnt)
#define ASYNC_LDS 1
typedef int v4i __attribute__((ext_vector_type(4)));
typedef __attribute__((address_space(1))) v4i GV4;   // global 16B chunk
typedef __attribute__((address_space(3))) v4i LV4;   // LDS 16B chunk
#else
#define ASYNC_LDS 0
#endif

// ---------------- degree / pseudo-coord prep ----------------

__global__ void deg_kernel(const int* __restrict__ dst, float* __restrict__ deg, int E) {
    int e = blockIdx.x * 256 + threadIdx.x;
    if (e < E) atomicAdd(&deg[dst[e]], 1.0f);
}

__global__ void rsq_kernel(const float* __restrict__ deg, float* __restrict__ rsq, int N) {
    int n = blockIdx.x * 256 + threadIdx.x;
    if (n < N) rsq[n] = rsqrtf(deg[n] + 1.0f);
}

// ---------------- embedding gather (writes f32 h and f16 shadow) ----------------

__global__ void embed_kernel(const int* __restrict__ idx, const float* __restrict__ emb,
                             float* __restrict__ h, _Float16* __restrict__ h16, int N) {
    int i = blockIdx.x * 256 + threadIdx.x;
    if (i < N * HID) {
        int n = i >> 6;
        float v = emb[idx[n] * HID + (i & 63)];
        h[i] = v;
        h16[i] = (_Float16)v;
    }
}

// ---------------- fp32 -> fp16 weight conversion ----------------

__global__ void convw_kernel(const float* __restrict__ w, _Float16* __restrict__ o, int n) {
    int i = blockIdx.x * 256 + threadIdx.x;
    if (i < n) o[i] = (_Float16)w[i];
}

// ---------------- WMMA GEMM: hp[N,192] = h16[N,64] @ W[192,64]^T ----------------
// 256 threads = 8 waves; each wave computes a 16-row strip across 12 col tiles.
// fp16 operands staged via async global->LDS copies (ASYNCcnt) on full blocks.

#define SW_STRIDE 72   // halves per row: 144B (16B aligned, conflict-friendly)

__global__ __launch_bounds__(256) void gemm_hp_kernel(const _Float16* __restrict__ h16,
                                                      const _Float16* __restrict__ Wf16,
                                                      float* __restrict__ hp, int N) {
    __shared__ _Float16 sA16[128 * SW_STRIDE];  // 18432 B
    __shared__ _Float16 sW[192 * SW_STRIDE];    // 27648 B

    const int tid  = threadIdx.x;
    const int wave = tid >> 5;
    const int lane = tid & 31;
    const int row0 = blockIdx.x * 128;
    const bool full = (row0 + 128 <= N);

#if ASYNC_LDS
    if (full) {
        // weights: 192 rows x 8 chunks of 8 halves (16B)
        for (int i = tid; i < 192 * 8; i += 256) {
            int j = i >> 3, kc = (i & 7) << 3;
            __builtin_amdgcn_global_load_async_to_lds_b128(
                (GV4*)(Wf16 + j * 64 + kc), (LV4*)(&sW[j * SW_STRIDE + kc]), 0, 0);
        }
        // h16 tile: 128 rows x 8 chunks of 8 halves (16B)
        for (int i = tid; i < 128 * 8; i += 256) {
            int r = i >> 3, kc = (i & 7) << 3;
            __builtin_amdgcn_global_load_async_to_lds_b128(
                (GV4*)(h16 + (row0 + r) * 64 + kc), (LV4*)(&sA16[r * SW_STRIDE + kc]), 0, 0);
        }
        __builtin_amdgcn_s_wait_asynccnt(0);
    } else
#endif
    {
        for (int i = tid; i < 192 * 64; i += 256)
            sW[(i >> 6) * SW_STRIDE + (i & 63)] = Wf16[i];
        for (int i = tid; i < 128 * 64; i += 256) {
            int r = i >> 6, c = i & 63;
            int gr = row0 + r;
            sA16[r * SW_STRIDE + c] = (gr < N) ? h16[gr * 64 + c] : (_Float16)0.0f;
        }
    }
    __syncthreads();

    const int half = lane >> 4;     // K sub-group for A / K group for B / row-half of D
    const int m    = lane & 15;     // A row within tile / B+D column

    // Build A fragments for both K=32 chunks (16x32 f16 layout per ISA)
    const _Float16* arow = &sA16[(wave * 16 + m) * SW_STRIDE];
    v16h afrag[2];
#pragma unroll
    for (int kk = 0; kk < 2; ++kk) {
#pragma unroll
        for (int v = 0; v < 8; ++v) {
            int kb = kk * 32 + ((v >> 2) << 4) + (half << 3) + ((v & 3) << 1);
            afrag[kk][2 * v]     = arow[kb];
            afrag[kk][2 * v + 1] = arow[kb + 1];
        }
    }

    const int rowbase = row0 + wave * 16 + (half << 3);   // first D row for this lane
    float* __restrict__ pbase = hp + rowbase * 192 + m;

#pragma unroll
    for (int tn = 0; tn < 12; ++tn) {
        v16h bfrag[2];
#pragma unroll
        for (int kk = 0; kk < 2; ++kk) {
#pragma unroll
            for (int v = 0; v < 8; ++v) {
                int j = tn * 16 + m;                      // output column (row of W)
                int k = kk * 32 + ((v >> 2) << 4) + (half << 3) + ((v & 3) << 1); // contraction index
                bfrag[kk][2 * v]     = sW[j * SW_STRIDE + k];
                bfrag[kk][2 * v + 1] = sW[j * SW_STRIDE + k + 1];
            }
        }
        v8f acc = {};
        acc = __builtin_amdgcn_wmma_f32_16x16x32_f16(false, afrag[0], false, bfrag[0],
                                                     (short)0, acc, false, false);
        acc = __builtin_amdgcn_wmma_f32_16x16x32_f16(false, afrag[1], false, bfrag[1],
                                                     (short)0, acc, false, false);
        if (full) {
            float* p = pbase + tn * 16;
#pragma unroll
            for (int r = 0; r < 8; ++r) p[r * 192] = acc[r];   // immediate offsets
        } else {
#pragma unroll
            for (int r = 0; r < 8; ++r) {
                int gm = rowbase + r;
                if (gm < N) hp[gm * 192 + tn * 16 + m] = acc[r];
            }
        }
    }
}

// ---------------- edge message + scatter-add (one wave per edge) ----------------

__global__ __launch_bounds__(256) void edge_kernel(const int* __restrict__ src,
                                                   const int* __restrict__ dst,
                                                   const float* __restrict__ rsq,
                                                   const float* __restrict__ hp,
                                                   float* __restrict__ agg,
                                                   const float* __restrict__ ppw,  // [2,2]
                                                   const float* __restrict__ ppb,  // [2]
                                                   const float* __restrict__ mu,   // [K,2]
                                                   const float* __restrict__ isig, // [K,2]
                                                   int E) {
    int e = blockIdx.x * 8 + (threadIdx.x >> 5);
    if (e >= E) return;
    const int lane = threadIdx.x & 31;
    const int s = src[e], d = dst[e];
    const float p0 = rsq[s], p1 = rsq[d];
    const float ps0 = tanhf(p0 * ppw[0] + p1 * ppw[1] + ppb[0]);
    const float ps1 = tanhf(p0 * ppw[2] + p1 * ppw[3] + ppb[1]);
    float g[KK];
#pragma unroll
    for (int k = 0; k < KK; ++k) {
        float d0 = ps0 - mu[k * 2 + 0], d1 = ps1 - mu[k * 2 + 1];
        float s0 = isig[k * 2 + 0],      s1 = isig[k * 2 + 1];
        g[k] = expf(-0.5f * (d0 * d0 * s0 * s0 + d1 * d1 * s1 * s1));
    }
    const float* hps = hp + s * 192;
    float* aggd = agg + d * 64;
#pragma unroll
    for (int u = 0; u < 2; ++u) {
        int c = lane + u * 32;
        float v = g[0] * hps[c] + g[1] * hps[64 + c] + g[2] * hps[128 + c];
        atomicAdd(&aggd[c], v);
    }
}

// ---------------- batchnorm stats (sum / sumsq per channel) ----------------

__global__ __launch_bounds__(256) void bn_stats_kernel(const float* __restrict__ agg,
                                                       float* __restrict__ stats, int N) {
    __shared__ float ssum[64], ssq[64];
    const int tid = threadIdx.x;
    if (tid < 64) { ssum[tid] = 0.0f; ssq[tid] = 0.0f; }
    __syncthreads();
    const int c = tid & 63;          // stride is multiple of 64 -> channel fixed per thread
    float s = 0.0f, q = 0.0f;
    const int total = N * 64;
    const int stride = gridDim.x * 256;
    for (int i = blockIdx.x * 256 + tid; i < total; i += stride) {
        float v = agg[i];
        s += v; q += v * v;
    }
    atomicAdd(&ssum[c], s);
    atomicAdd(&ssq[c], q);
    __syncthreads();
    if (tid < 64) {
        atomicAdd(&stats[tid], ssum[tid]);
        atomicAdd(&stats[64 + tid], ssq[tid]);
    }
}

// ---------------- batchnorm apply + relu + residual (writes f32 + f16) ----------------

__global__ void bn_apply_kernel(const float* __restrict__ agg, const float* __restrict__ stats,
                                const float* __restrict__ gma, const float* __restrict__ bta,
                                float* __restrict__ h, _Float16* __restrict__ h16, int N) {
    int i = blockIdx.x * 256 + threadIdx.x;
    if (i >= N * 64) return;
    int c = i & 63;
    float invN = 1.0f / (float)N;
    float mean = stats[c] * invN;
    float var  = stats[64 + c] * invN - mean * mean;
    float hn = (agg[i] - mean) * rsqrtf(var + BN_EPS) * gma[c] + bta[c];
    float nh = h[i] + fmaxf(hn, 0.0f);
    h[i] = nh;
    h16[i] = (_Float16)nh;
}

// ---------------- fused MLP readout: 64 -> 32 -> 16 -> 7 ----------------

__global__ __launch_bounds__(256) void readout_kernel(const float* __restrict__ h,
                                                      const float* __restrict__ w1, const float* __restrict__ b1,
                                                      const float* __restrict__ w2, const float* __restrict__ b2,
                                                      const float* __restrict__ w3, const float* __restrict__ b3,
                                                      float* __restrict__ out, int N) {
    __shared__ float sw1[32 * 64], sw2[16 * 32], sw3[7 * 16];
    __shared__ float sb1[32], sb2[16], sb3[7];
    const int tid = threadIdx.x;
    for (int i = tid; i < 32 * 64; i += 256) sw1[i] = w1[i];
    for (int i = tid; i < 16 * 32; i += 256) sw2[i] = w2[i];
    for (int i = tid; i < 7 * 16;  i += 256) sw3[i] = w3[i];
    if (tid < 32) sb1[tid] = b1[tid];
    if (tid < 16) sb2[tid] = b2[tid];
    if (tid < 7)  sb3[tid] = b3[tid];
    __syncthreads();
    int n = blockIdx.x * 256 + tid;
    if (n >= N) return;
    float hv[64];
    const float* hn = h + n * 64;
#pragma unroll
    for (int c = 0; c < 64; ++c) hv[c] = hn[c];
    float y1[32];
#pragma unroll
    for (int j = 0; j < 32; ++j) {
        float a = sb1[j];
#pragma unroll
        for (int c = 0; c < 64; ++c) a += sw1[j * 64 + c] * hv[c];
        y1[j] = fmaxf(a, 0.0f);
    }
    float y2[16];
#pragma unroll
    for (int j = 0; j < 16; ++j) {
        float a = sb2[j];
#pragma unroll
        for (int c = 0; c < 32; ++c) a += sw2[j * 32 + c] * y1[c];
        y2[j] = fmaxf(a, 0.0f);
    }
#pragma unroll
    for (int j = 0; j < NC; ++j) {
        float a = sb3[j];
#pragma unroll
        for (int c = 0; c < 16; ++c) a += sw3[j * 16 + c] * y2[c];
        out[n * NC + j] = a;
    }
}

// ---------------- host launcher ----------------

extern "C" void kernel_launch(void* const* d_in, const int* in_sizes, int n_in,
                              void* d_out, int out_size, void* d_ws, size_t ws_size,
                              hipStream_t stream) {
    const int*   h_idx = (const int*)  d_in[0];
    const int*   eidx  = (const int*)  d_in[1];
    const float* emb   = (const float*)d_in[2];
    const float* fc_w  = (const float*)d_in[3];
    const float* mu    = (const float*)d_in[4];
    const float* isig  = (const float*)d_in[5];
    const float* bn_g  = (const float*)d_in[6];
    const float* bn_b  = (const float*)d_in[7];
    const float* pp_w  = (const float*)d_in[8];
    const float* pp_b  = (const float*)d_in[9];
    const float* w1    = (const float*)d_in[10];
    const float* b1    = (const float*)d_in[11];
    const float* w2    = (const float*)d_in[12];
    const float* b2    = (const float*)d_in[13];
    const float* w3    = (const float*)d_in[14];
    const float* b3    = (const float*)d_in[15];
    float* out = (float*)d_out;

    const int N = in_sizes[0];
    const int E = in_sizes[1] / 2;
    const int* src = eidx;
    const int* dst = eidx + E;

    // carve workspace
    char* ws = (char*)d_ws;
    size_t off = 0;
    auto carve = [&](size_t bytes) -> void* {
        off = (off + 255) & ~(size_t)255;
        void* p = ws + off;
        off += bytes;
        return p;
    };
    float*    d_deg   = (float*)   carve((size_t)N * 4);
    float*    d_rsq   = (float*)   carve((size_t)N * 4);
    float*    d_h     = (float*)   carve((size_t)N * HID * 4);
    _Float16* d_h16   = (_Float16*)carve((size_t)N * HID * 2);
    float*    d_hp    = (float*)   carve((size_t)N * KK * HID * 4);
    float*    d_agg   = (float*)   carve((size_t)N * HID * 4);
    float*    d_stats = (float*)   carve(128 * 4);
    _Float16* d_wf16  = (_Float16*)carve((size_t)NL * KK * HID * HID * 2);
    (void)ws_size; (void)n_in; (void)out_size;

    // degrees + rsqrt(deg+1)
    (void)hipMemsetAsync(d_deg, 0, (size_t)N * 4, stream);
    deg_kernel<<<(E + 255) / 256, 256, 0, stream>>>(dst, d_deg, E);
    rsq_kernel<<<(N + 255) / 256, 256, 0, stream>>>(d_deg, d_rsq, N);

    // fp16 weights + embedding
    const int nw = NL * KK * HID * HID;
    convw_kernel<<<(nw + 255) / 256, 256, 0, stream>>>(fc_w, d_wf16, nw);
    embed_kernel<<<(N * HID + 255) / 256, 256, 0, stream>>>(h_idx, emb, d_h, d_h16, N);

    const int gemm_blocks = (N + 127) / 128;
    for (int l = 0; l < NL; ++l) {
        gemm_hp_kernel<<<gemm_blocks, 256, 0, stream>>>(
            d_h16, d_wf16 + (size_t)l * KK * HID * HID, d_hp, N);

        (void)hipMemsetAsync(d_agg, 0, (size_t)N * HID * 4, stream);
        edge_kernel<<<(E + 7) / 8, 256, 0, stream>>>(
            src, dst, d_rsq, d_hp, d_agg,
            pp_w + l * 4, pp_b + l * 2, mu + l * KK * 2, isig + l * KK * 2, E);

        (void)hipMemsetAsync(d_stats, 0, 128 * 4, stream);
        bn_stats_kernel<<<256, 256, 0, stream>>>(d_agg, d_stats, N);
        bn_apply_kernel<<<(N * HID + 255) / 256, 256, 0, stream>>>(
            d_agg, d_stats, bn_g + l * HID, bn_b + l * HID, d_h, d_h16, N);
    }

    readout_kernel<<<(N + 255) / 256, 256, 0, stream>>>(
        d_h, w1, b1, w2, b2, w3, b3, out, N);
}
